// WinDecoderTransformer_75230647157178
// MI455X (gfx1250) — matrix operand order repaired and checked by reading, not compiled, not run on hardware
//
#include <hip/hip_runtime.h>
#include <hip/hip_bf16.h>
#include <math.h>

typedef __attribute__((ext_vector_type(16))) __bf16 v16bf;
typedef __attribute__((ext_vector_type(8)))  __bf16 v8bf;
typedef __attribute__((ext_vector_type(8)))  float  v8f;

static constexpr int kBW    = 512;                 // windows: B=4 * 16*8
static constexpr int kC     = 256;                 // channels
static constexpr int kN     = 128;                 // spatial per window (8x16)
static constexpr int kWElem = kBW * kC * kN;       // 16,777,216
static constexpr int kL     = 2;

// ---------------- bf16 helpers ----------------
__device__ __forceinline__ unsigned short f2bf_bits(float f) {
  union { float f; unsigned u; } v; v.f = f;
  unsigned r = v.u + 0x7FFFu + ((v.u >> 16) & 1u);   // round to nearest even
  return (unsigned short)(r >> 16);
}
__device__ __forceinline__ __bf16 bf_from_bits(unsigned short h) {
  return __builtin_bit_cast(__bf16, h);
}
__device__ __forceinline__ v16bf cat8(v8bf a, v8bf b) {
  return __builtin_shufflevector(a, b, 0,1,2,3,4,5,6,7,8,9,10,11,12,13,14,15);
}
__device__ __forceinline__ v8f wmma_bf16(v16bf a, v16bf b, v8f c) {
  return __builtin_amdgcn_wmma_f32_16x16x32_bf16(false, a, false, b,
                                                 (short)0, c, false, false);
}

// ---- CDNA5 async global->LDS copy (ASYNCcnt path, CDNA5 ISA) ----
__device__ __forceinline__ void async_load_b128(const void* gptr, void* lptr) {
  unsigned long long ga = (unsigned long long)gptr;
  unsigned la = (unsigned)(unsigned long long)lptr;   // low 32 bits = LDS offset
  asm volatile("global_load_async_to_lds_b128 %0, %1, off"
               :: "v"(la), "v"(ga) : "memory");
}
__device__ __forceinline__ void wait_async0() {
  asm volatile("s_wait_asynccnt 0" ::: "memory");
}

// ---------------- window partition: src->mem(bf16), query->tgt(f32) ----------------
__global__ void k_winpart(const float* __restrict__ src,
                          const float* __restrict__ qry,
                          unsigned short* __restrict__ membf,
                          float* __restrict__ tgt) {
  int idx = blockIdx.x * 256 + threadIdx.x;
  int wi  = idx >> 15;
  int rem = idx & 32767;
  int c   = rem >> 7;
  int p   = rem & 127;
  int b   = wi >> 7;
  int r   = wi & 127;
  int y   = (r >> 3) * 8  + (p >> 4);
  int x   = (r & 7)  * 16 + (p & 15);
  size_t s = (((size_t)b * kC + c) * 128 + y) * 128 + x;
  membf[idx] = f2bf_bits(src[s]);
  tgt[idx]   = qry[s];
}

// ---------------- per-window mean / rsqrt(var+eps) ----------------
__global__ void k_stats(const float* __restrict__ x,
                        float* __restrict__ mu, float* __restrict__ rs) {
  int wi = blockIdx.x, tid = threadIdx.x;
  size_t off = (size_t)wi * 32768;
  float s = 0.f, sq = 0.f;
  for (int i = tid; i < 32768; i += 256) { float v = x[off + i]; s += v; sq += v * v; }
  __shared__ float ss[256], sqq[256];
  ss[tid] = s; sqq[tid] = sq; __syncthreads();
  for (int st = 128; st > 0; st >>= 1) {
    if (tid < st) { ss[tid] += ss[tid + st]; sqq[tid] += sqq[tid + st]; }
    __syncthreads();
  }
  if (tid == 0) {
    float m = ss[0] * (1.f / 32768.f);
    float var = sqq[0] * (1.f / 32768.f) - m * m;
    mu[wi] = m; rs[wi] = rsqrtf(var + 1e-5f);
  }
}

// ---------------- f32 -> bf16 bits (weights) ----------------
__global__ void k_f32bf(const float* __restrict__ in,
                        unsigned short* __restrict__ out, int n) {
  int i = blockIdx.x * 256 + threadIdx.x;
  if (i < n) out[i] = f2bf_bits(in[i]);
}

// -------- fused gn+avgpool mix: out = base + ls*gnw*rs*(pool(x)-x); optional bf16(x) --------
__global__ void k_mix(const float* __restrict__ x, const float* __restrict__ basep,
                      float* __restrict__ out, const float* __restrict__ rs,
                      const float* __restrict__ lsv, const float* __restrict__ gnw,
                      unsigned short* __restrict__ xbf) {
  int blk = blockIdx.x;
  int wi  = blk >> 8;
  int c   = blk & 255;
  int t   = threadIdx.x;              // 0..127, one 8x16 channel tile
  __shared__ float tile[128];
  size_t o = (size_t)wi * 32768 + (size_t)c * 128;
  float v = x[o + t];
  tile[t] = v;
  if (xbf) xbf[o + t] = f2bf_bits(v);
  __syncthreads();
  int y = t >> 4, xx = t & 15;
  int y0 = y  > 0  ? y  - 1 : 0,  y1 = y  < 7  ? y  + 1 : 7;
  int x0 = xx > 0  ? xx - 1 : 0,  x1 = xx < 15 ? xx + 1 : 15;
  float s = 0.f;
  for (int yy = y0; yy <= y1; ++yy)
    for (int xc = x0; xc <= x1; ++xc) s += tile[yy * 16 + xc];
  float pool = s / (float)((y1 - y0 + 1) * (x1 - x0 + 1));
  out[o + t] = basep[o + t] + lsv[c] * gnw[c] * rs[wi] * (pool - v);
}

// ---------------- full GroupNorm affine -> bf16 ----------------
__global__ void k_gnaff(const float* __restrict__ x, const float* __restrict__ mu,
                        const float* __restrict__ rs, const float* __restrict__ w,
                        const float* __restrict__ bb,
                        unsigned short* __restrict__ out) {
  int idx = blockIdx.x * 256 + threadIdx.x;
  int wi = idx >> 15;
  int c  = (idx >> 7) & 255;
  out[idx] = f2bf_bits((x[idx] - mu[wi]) * rs[wi] * w[c] + bb[c]);
}

// ---------------- batched GEMM: out[wi][m][p] = epi(sum_k W[m][k]*X[wi][k][p] + bias[m])
// MODE 0: bf16.  MODE 1: GELU(erf)->bf16.  MODE 2: f32 resid + ls*val.  MODE 3: bf16 V-transpose.
template <int MODE>
__global__ void k_gemm(const unsigned short* __restrict__ Wb,
                       const unsigned short* __restrict__ Xb,
                       const float* __restrict__ bias,
                       void* __restrict__ outp,
                       const float* __restrict__ resid,
                       const float* __restrict__ lsv,
                       int M, int K) {
  const int wi   = blockIdx.x;
  const int m0   = blockIdx.y * 128;
  const int tid  = threadIdx.x;
  const int wave = tid >> 5;
  const int lane = tid & 31;
  __shared__ __align__(16) unsigned short As[128][40];   // [m][k], 80B row (16B mult)
  __shared__ __align__(16) unsigned short Bs[32][136];   // [k][n], 272B row (16B mult)
  const v8f vz = {0.f,0.f,0.f,0.f,0.f,0.f,0.f,0.f};
  v8f acc[8];
#pragma unroll
  for (int t = 0; t < 8; ++t) acc[t] = vz;
  const size_t xbase = (size_t)wi * (size_t)K * kN;

  for (int k0 = 0; k0 < K; k0 += 32) {
    // async global->LDS staging: 8KB per tile, 2 b128 issues per thread per tile
#pragma unroll
    for (int cix = 0; cix < 2; ++cix) {
      int ch = tid + cix * 256;                 // 0..511 chunks of 8 ushorts
      int ar = ch >> 2, ac = (ch & 3) * 8;      // A: 4 chunks per 32-wide row
      async_load_b128(&Wb[(size_t)(m0 + ar) * K + (k0 + ac)], &As[ar][ac]);
      int br = ch >> 4, bc = (ch & 15) * 8;     // B: 16 chunks per 128-wide row
      async_load_b128(&Xb[xbase + (size_t)(k0 + br) * kN + bc], &Bs[br][bc]);
    }
    if (k0 + 32 < K) {   // global_prefetch_b8 for next K chunk
      __builtin_prefetch(&Xb[xbase + (size_t)(k0 + 32) * kN + tid], 0, 0);
      __builtin_prefetch(&Wb[(size_t)(m0 + (tid >> 1)) * K + k0 + 32], 0, 0);
    }
    wait_async0();
    __syncthreads();
    // A fragment: 16x32 strip for this wave (contiguous k = 16*(lane>>4) + s, same as B), 2x ds_load_b128
    v16bf afr;
    {
      int row = wave * 16 + (lane & 15);
      int ko  = (lane >> 4) * 16;
      v8bf a0 = *(const v8bf*)&As[row][ko];
      v8bf a1 = *(const v8bf*)&As[row][ko + 8];
      afr = cat8(a0, a1);
    }
#pragma unroll
    for (int nt = 0; nt < 8; ++nt) {
      v16bf bfr;
      int col = nt * 16 + (lane & 15);
      int kb  = (lane >> 4) * 16;
#pragma unroll
      for (int j = 0; j < 16; ++j) bfr[j] = bf_from_bits(Bs[kb + j][col]);
      acc[nt] = wmma_bf16(afr, bfr, acc[nt]);
    }
    __syncthreads();
  }
  // epilogue (C/D layout: row = (lane>>4)*8 + r, col = lane&15)
#pragma unroll
  for (int nt = 0; nt < 8; ++nt) {
#pragma unroll
    for (int r = 0; r < 8; ++r) {
      int gm = m0 + wave * 16 + (lane >> 4) * 8 + r;
      int gn = nt * 16 + (lane & 15);
      float v = acc[nt][r] + bias[gm];
      if (MODE == 0) {
        size_t o = (size_t)wi * (size_t)M * kN + (size_t)gm * kN + gn;
        ((unsigned short*)outp)[o] = f2bf_bits(v);
      } else if (MODE == 1) {
        size_t o = (size_t)wi * (size_t)M * kN + (size_t)gm * kN + gn;
        float g = 0.5f * v * (1.0f + erff(v * 0.70710678118f));
        ((unsigned short*)outp)[o] = f2bf_bits(g);
      } else if (MODE == 2) {
        size_t o = (size_t)wi * (size_t)M * kN + (size_t)gm * kN + gn;
        ((float*)outp)[o] = resid[o] + lsv[gm] * v;
      } else { // MODE 3: store token-transposed Vt[feat][token] for attention B-frags
        size_t o = (size_t)wi * 32768 +
                   (size_t)((gm & 1) * 128 + gn) * 128 + (size_t)(gm >> 1);
        ((unsigned short*)outp)[o] = f2bf_bits(v);
      }
    }
  }
}

// ---------------- per-window attention: S=softmax(Qm Km^T); out = tgt2 + S Vm ----------------
// Qm/Km are raw-flat [128 tok][256 feat] views; Vt is transposed [256 feat][128 tok].
__global__ void k_attn(const unsigned short* __restrict__ Q,
                       const unsigned short* __restrict__ Kd,
                       const unsigned short* __restrict__ Vt,
                       const float* __restrict__ tgt2,
                       float* __restrict__ outT) {
  const int wi   = blockIdx.x;
  const int tid  = threadIdx.x;
  const int wave = tid >> 5;
  const int lane = tid & 31;
  __shared__ float S[128][128];                 // 64 KB LDS
  const size_t base = (size_t)wi * 32768;

  // Phase 1: wave's 16-row Q strip, A fragments for all 8 K-chunks live in regs
  v16bf afr[8];
  {
    int row = wave * 16 + (lane & 15);
    int ko  = (lane >> 4) * 16;
    const unsigned short* qrow = Q + base + (size_t)row * 256;
#pragma unroll
    for (int c = 0; c < 8; ++c) {
      v8bf a0 = *(const v8bf*)(qrow + c * 32 + ko);
      v8bf a1 = *(const v8bf*)(qrow + c * 32 + ko + 8);
      afr[c] = cat8(a0, a1);
    }
  }
  const v8f vz = {0.f,0.f,0.f,0.f,0.f,0.f,0.f,0.f};
  for (int mt = 0; mt < 8; ++mt) {              // key tiles
    v8f acc = vz;
    int mrow = mt * 16 + (lane & 15);
    const unsigned short* krow = Kd + base + (size_t)mrow * 256 + (lane >> 4) * 16;
#pragma unroll
    for (int c = 0; c < 8; ++c) {               // B[kk][nn] = Km[n][k], contiguous 16
      v16bf bfr = *(const v16bf*)(krow + c * 32);
      acc = wmma_bf16(afr[c], bfr, acc);
    }
#pragma unroll
    for (int r = 0; r < 8; ++r)
      S[wave * 16 + (lane >> 4) * 8 + r][mt * 16 + (lane & 15)] = acc[r];
  }
  __syncthreads();

  // Row softmax (no scale factor in reference)
  if (tid < 128) {
    float mx = -3.4e38f;
    for (int j = 0; j < 128; ++j) mx = fmaxf(mx, S[tid][j]);
    float s = 0.f;
    for (int j = 0; j < 128; ++j) { float e = __expf(S[tid][j] - mx); S[tid][j] = e; s += e; }
    float inv = 1.f / s;
    for (int j = 0; j < 128; ++j) S[tid][j] *= inv;
  }
  __syncthreads();

  // Phase 2: O = P @ Vm, fused residual add; Vt gives contiguous B-frags
  for (int ct = 0; ct < 16; ++ct) {             // 16 feature tiles of 16
    v8f acc = vz;
#pragma unroll
    for (int c4 = 0; c4 < 4; ++c4) {            // K = 128 tokens in chunks of 32
      v16bf pa;
      {
        int row = wave * 16 + (lane & 15);
        int ko  = (lane >> 4) * 16;
        const float* sp = &S[row][c4 * 32];
#pragma unroll
        for (int i = 0; i < 8; ++i) {
          pa[i]     = bf_from_bits(f2bf_bits(sp[ko + i]));
          pa[i + 8] = bf_from_bits(f2bf_bits(sp[ko + 8 + i]));
        }
      }
      int col = ct * 16 + (lane & 15);          // feature
      int kb  = c4 * 32 + (lane >> 4) * 16;     // token base
      v16bf bv = *(const v16bf*)(Vt + base + (size_t)col * 128 + kb);
      acc = wmma_bf16(pa, bv, acc);
    }
#pragma unroll
    for (int r = 0; r < 8; ++r) {
      int row = wave * 16 + (lane >> 4) * 8 + r;
      int col = ct * 16 + (lane & 15);
      size_t o = base + (size_t)row * 256 + col;
      outT[o] = tgt2[o] + acc[r];               // t = tgt2 + guided
    }
  }
}

// ---------------- final GN + window reverse into [1, HW, B, C] ----------------
__global__ void k_final(const float* __restrict__ x, const float* __restrict__ mu,
                        const float* __restrict__ rs, const float* __restrict__ gw,
                        const float* __restrict__ gb, float* __restrict__ out) {
  int idx = blockIdx.x * 256 + threadIdx.x;
  int wi  = idx >> 15;
  int rem = idx & 32767;
  int c   = rem >> 7;
  int p   = rem & 127;
  int b   = wi >> 7;
  int r   = wi & 127;
  int y   = (r >> 3) * 8  + (p >> 4);
  int xx  = (r & 7)  * 16 + (p & 15);
  int n   = y * 128 + xx;
  float v = (x[idx] - mu[wi]) * rs[wi] * gw[c] + gb[c];
  out[((size_t)n * 4 + b) * 256 + c] = v;
}

// =======================================================================
extern "C" void kernel_launch(void* const* d_in, const int* in_sizes, int n_in,
                              void* d_out, int out_size, void* d_ws, size_t ws_size,
                              hipStream_t stream) {
  (void)in_sizes; (void)n_in; (void)out_size; (void)ws_size;
  const float* src  = (const float*)d_in[0];
  const float* qry  = (const float*)d_in[1];
  const float* gn1w = (const float*)d_in[2];
  const float* gn2w = (const float*)d_in[4];
  const float* gn2b = (const float*)d_in[5];
  const float* ls1  = (const float*)d_in[6];
  const float* ls2  = (const float*)d_in[7];
  const float* ls3  = (const float*)d_in[8];
  const float* qw   = (const float*)d_in[9];
  const float* qb   = (const float*)d_in[10];
  const float* kw   = (const float*)d_in[11];
  const float* kb   = (const float*)d_in[12];
  const float* vw   = (const float*)d_in[13];
  const float* vb   = (const float*)d_in[14];
  const float* f1w  = (const float*)d_in[15];
  const float* f1b  = (const float*)d_in[16];
  const float* f2w  = (const float*)d_in[17];
  const float* f2b  = (const float*)d_in[18];
  const float* gfw  = (const float*)d_in[19];
  const float* gfb  = (const float*)d_in[20];

  char* ws = (char*)d_ws;
  size_t off = 0;
  auto alloc = [&](size_t bytes) {
    void* p = ws + off;
    off += (bytes + 255) & ~(size_t)255;
    return p;
  };
  float* tgtA  = (float*)alloc((size_t)kWElem * 4);   // residual stream (layer in / fc2 out)
  float* tgt2  = (float*)alloc((size_t)kWElem * 4);
  float* tBuf  = (float*)alloc((size_t)kWElem * 4);   // tgt2 + guided
  float* tgt3  = (float*)alloc((size_t)kWElem * 4);
  unsigned short* membf = (unsigned short*)alloc((size_t)kWElem * 2);
  unsigned short* tbf   = (unsigned short*)alloc((size_t)kWElem * 2);
  unsigned short* qbuf  = (unsigned short*)alloc((size_t)kWElem * 2);
  unsigned short* kbuf  = (unsigned short*)alloc((size_t)kWElem * 2);
  unsigned short* vbuf  = (unsigned short*)alloc((size_t)kWElem * 2);  // transposed Vt
  unsigned short* gnbf  = (unsigned short*)alloc((size_t)kWElem * 2);
  unsigned short* h1bf  = (unsigned short*)alloc((size_t)kBW * 512 * kN * 2);
  unsigned short* wqbf  = (unsigned short*)alloc((size_t)kL * kC * kC * 2);
  unsigned short* wkbf  = (unsigned short*)alloc((size_t)kL * kC * kC * 2);
  unsigned short* wvbf  = (unsigned short*)alloc((size_t)kL * kC * kC * 2);
  unsigned short* f1bf  = (unsigned short*)alloc((size_t)kL * 512 * kC * 2);
  unsigned short* f2bf  = (unsigned short*)alloc((size_t)kL * kC * 512 * 2);
  float* mu = (float*)alloc(kBW * 4);
  float* rs = (float*)alloc(kBW * 4);

  const int EB = kWElem / 256;   // 65536 elementwise blocks

  // window partition + bf16 weights
  k_winpart<<<EB, 256, 0, stream>>>(src, qry, membf, tgtA);
  k_f32bf<<<(kL*kC*kC + 255)/256, 256, 0, stream>>>(qw, wqbf, kL*kC*kC);
  k_f32bf<<<(kL*kC*kC + 255)/256, 256, 0, stream>>>(kw, wkbf, kL*kC*kC);
  k_f32bf<<<(kL*kC*kC + 255)/256, 256, 0, stream>>>(vw, wvbf, kL*kC*kC);
  k_f32bf<<<(kL*512*kC + 255)/256, 256, 0, stream>>>(f1w, f1bf, kL*512*kC);
  k_f32bf<<<(kL*kC*512 + 255)/256, 256, 0, stream>>>(f2w, f2bf, kL*kC*512);

  for (int l = 0; l < kL; ++l) {
    // tgt2 = tgt + ls1*gn1w*rs*(pool(tgt) - tgt)   [mean & bias cancel]; also bf16(tgt)
    k_stats<<<kBW, 256, 0, stream>>>(tgtA, mu, rs);
    k_mix<<<kBW * kC, 128, 0, stream>>>(tgtA, tgtA, tgt2, rs, ls1 + l*kC, gn1w, tbf);

    // Q from tgt, K/V from mem (V stored transposed)
    k_gemm<0><<<dim3(kBW, kC/128), 256, 0, stream>>>(wqbf + (size_t)l*kC*kC, tbf,
        qb + l*kC, qbuf, nullptr, nullptr, kC, kC);
    k_gemm<0><<<dim3(kBW, kC/128), 256, 0, stream>>>(wkbf + (size_t)l*kC*kC, membf,
        kb + l*kC, kbuf, nullptr, nullptr, kC, kC);
    k_gemm<3><<<dim3(kBW, kC/128), 256, 0, stream>>>(wvbf + (size_t)l*kC*kC, membf,
        vb + l*kC, vbuf, nullptr, nullptr, kC, kC);

    // t = tgt2 + AGP(tgt, mem)
    k_attn<<<kBW, 256, 0, stream>>>(qbuf, kbuf, vbuf, tgt2, tBuf);

    // tgt3 = tgt2 + ls2*gn1w*rs(t)*(pool(t) - t)
    k_stats<<<kBW, 256, 0, stream>>>(tBuf, mu, rs);
    k_mix<<<kBW * kC, 128, 0, stream>>>(tBuf, tgt2, tgt3, rs, ls2 + l*kC, gn1w, nullptr);

    // MLP: gn2 -> fc1 -> gelu -> fc2, residual with ls3
    k_stats<<<kBW, 256, 0, stream>>>(tgt3, mu, rs);
    k_gnaff<<<EB, 256, 0, stream>>>(tgt3, mu, rs, gn2w + l*kC, gn2b + l*kC, gnbf);
    k_gemm<1><<<dim3(kBW, 512/128), 256, 0, stream>>>(f1bf + (size_t)l*512*kC, gnbf,
        f1b + l*512, h1bf, nullptr, nullptr, 512, kC);
    k_gemm<2><<<dim3(kBW, kC/128), 256, 0, stream>>>(f2bf + (size_t)l*kC*512, h1bf,
        f2b + l*kC, tgtA, tgt3, ls3 + l*kC, kC, 512);
  }

  // final GN + window reverse into [1, HW, B, C]
  k_stats<<<kBW, 256, 0, stream>>>(tgtA, mu, rs);
  k_final<<<EB, 256, 0, stream>>>(tgtA, mu, rs, gfw, gfb, (float*)d_out);
}
